// Transformer_41308995453485
// MI455X (gfx1250) — matrix operand order repaired and checked by reading, not compiled, not run on hardware
//
#include <hip/hip_runtime.h>

// ---------------- model dims ----------------
#define TB    2
#define TS    1024
#define TD    1024
#define TH    16
#define TKVH  4
#define THD   64
#define TE    8
#define TLAY  2
#define TV    32000
#define TI    3584
#define TT    (TB*TS)          // 2048 tokens
#define TQKV  1536             // (H + 2*KVH)*HD

typedef __attribute__((ext_vector_type(16))) __bf16 v16bf;
typedef __attribute__((ext_vector_type(8)))  float  v8f;

union FragBF { unsigned short u[16]; v16bf v; };

__device__ __forceinline__ unsigned short f32_bf16(float f) {
    unsigned int u = __float_as_uint(f);
    u += 0x7FFFu + ((u >> 16) & 1u);          // round-to-nearest-even
    return (unsigned short)(u >> 16);
}

// =====================================================================
// Generic GEMM: C[M,N] = epilogue( A[M,K] * W[N,K]^T )
// mode 0: C = acc ; mode 1: C = acc + aux ; mode 2: C = silu(aux)*acc
// Block tile 128x128, 8 waves (4x2), wave tile 32x64 (2x4 wmma tiles).
// =====================================================================
#define GM 128
#define GN 128
#define GK 32

__global__ __launch_bounds__(256)
void gemm_bf16(const float* __restrict__ A, const float* __restrict__ W,
               float* __restrict__ C, const float* __restrict__ aux,
               int M, int N, int Kd, int mode)
{
    __shared__ unsigned short As[GM][GK + 8];
    __shared__ unsigned short Bs[GN][GK + 8];

    const int tid  = threadIdx.x;
    const int lane = tid & 31;
    const int wave = tid >> 5;
    const int wm   = wave >> 1;       // 0..3
    const int wn   = wave & 1;        // 0..1
    const int hi   = lane >> 4;       // lane half
    const int l16  = lane & 15;
    const int row0 = blockIdx.y * GM;
    const int col0 = blockIdx.x * GN;

    v8f zero = {};
    v8f acc[2][4];
    #pragma unroll
    for (int i = 0; i < 2; ++i)
        #pragma unroll
        for (int j = 0; j < 4; ++j) acc[i][j] = zero;

    for (int k0 = 0; k0 < Kd; k0 += GK) {
        __syncthreads();
        #pragma unroll
        for (int i = 0; i < 16; ++i) {          // 128x32 A tile, f32 -> bf16
            int e = tid + i * 256;
            int r = e >> 5, k = e & 31;
            As[r][k] = f32_bf16(A[(size_t)(row0 + r) * Kd + k0 + k]);
        }
        #pragma unroll
        for (int i = 0; i < 16; ++i) {          // 128x32 W tile (N-major rows of K)
            int e = tid + i * 256;
            int r = e >> 5, k = e & 31;
            Bs[r][k] = f32_bf16(W[(size_t)(col0 + r) * Kd + k0 + k]);
        }
        __syncthreads();

        FragBF a[2], b[4];
        #pragma unroll
        for (int i = 0; i < 2; ++i) {           // A frag: lane m=l16, K octets by half
            int m = wm * 32 + i * 16 + l16;
            #pragma unroll
            for (int j = 0; j < 8; ++j) {
                a[i].u[j]     = As[m][hi * 8 + j];
                a[i].u[8 + j] = As[m][16 + hi * 8 + j];
            }
        }
        #pragma unroll
        for (int j4 = 0; j4 < 4; ++j4) {        // B frag: lane n=l16, K octets by half
            int n = wn * 64 + j4 * 16 + l16;
            #pragma unroll
            for (int j = 0; j < 8; ++j) { b[j4].u[j] = Bs[n][hi * 8 + j]; b[j4].u[8 + j] = Bs[n][16 + hi * 8 + j]; }
        }
        #pragma unroll
        for (int i = 0; i < 2; ++i)
            #pragma unroll
            for (int j = 0; j < 4; ++j)
                acc[i][j] = __builtin_amdgcn_wmma_f32_16x16x32_bf16(
                    false, a[i].v, false, b[j].v, (short)0, acc[i][j], false, false);
    }

    #pragma unroll
    for (int i = 0; i < 2; ++i)
        #pragma unroll
        for (int j = 0; j < 4; ++j)
            #pragma unroll
            for (int g = 0; g < 8; ++g) {
                int m = row0 + wm * 32 + i * 16 + hi * 8 + g;
                int n = col0 + wn * 64 + j * 16 + l16;
                size_t idx = (size_t)m * N + n;
                float v = acc[i][j][g];
                if (mode == 1) v += aux[idx];
                else if (mode == 2) { float a0 = aux[idx]; v *= a0 / (1.f + __expf(-a0)); }
                C[idx] = v;
            }
}

// =====================================================================
// Flash attention: one wave per (b, h, 16-row q tile); key chunks of 32.
// =====================================================================
__global__ __launch_bounds__(128)
void attn_kernel(const float* __restrict__ qkv, float* __restrict__ y)
{
    __shared__ unsigned short Pl[4][16][32];
    const int lane = threadIdx.x;
    const int wv   = threadIdx.y;
    int lin = blockIdx.x * 4 + wv;            // 0 .. B*H*64-1
    const int qt = lin & 63;  lin >>= 6;
    const int h  = lin & 15;  lin >>= 4;
    const int b  = lin;
    const int kvh = h >> 2;
    const int hi  = lane >> 4;
    const int l16 = lane & 15;

    FragBF qa[2];                              // q row, 1/sqrt(64) folded in
    {
        const int sq = qt * 16 + l16;
        const float* qp = qkv + (size_t)(b * TS + sq) * TQKV + h * THD;
        #pragma unroll
        for (int kw = 0; kw < 2; ++kw)
            #pragma unroll
            for (int j = 0; j < 8; ++j) {
                qa[kw].u[j]     = f32_bf16(qp[kw * 32 + hi * 8 + j] * 0.125f);
                qa[kw].u[8 + j] = f32_bf16(qp[kw * 32 + 16 + hi * 8 + j] * 0.125f);
            }
    }

    float mrun[8], lrun[8];
    #pragma unroll
    for (int g = 0; g < 8; ++g) { mrun[g] = -1e30f; lrun[g] = 0.f; }
    v8f zero = {};
    v8f o[4];
    #pragma unroll
    for (int j = 0; j < 4; ++j) o[j] = zero;

    const int nkc = (qt * 16 + 16 + 31) >> 5;
    for (int kc = 0; kc < nkc; ++kc) {
        v8f s0 = zero, s1 = zero;
        #pragma unroll
        for (int kw = 0; kw < 2; ++kw) {       // HD=64 => two K=32 wmma steps
            FragBF kb0, kb1;
            const int key0 = kc * 32 + l16;
            const float* kp0 = qkv + (size_t)(b * TS + key0) * TQKV + TD + kvh * THD + kw * 32 + hi * 8;
            const float* kp1 = kp0 + (size_t)16 * TQKV;
            #pragma unroll
            for (int j = 0; j < 8; ++j) { kb0.u[j] = f32_bf16(kp0[j]); kb1.u[j] = f32_bf16(kp1[j]); kb0.u[8 + j] = f32_bf16(kp0[16 + j]); kb1.u[8 + j] = f32_bf16(kp1[16 + j]); }
            s0 = __builtin_amdgcn_wmma_f32_16x16x32_bf16(false, qa[kw].v, false, kb0.v, (short)0, s0, false, false);
            s1 = __builtin_amdgcn_wmma_f32_16x16x32_bf16(false, qa[kw].v, false, kb1.v, (short)0, s1, false, false);
        }

        float p0a[8], p1a[8];
        #pragma unroll
        for (int g = 0; g < 8; ++g) {          // causal mask + online softmax
            const int sq   = qt * 16 + hi * 8 + g;
            const int key0 = kc * 32 + l16;
            float v0 = (key0      <= sq) ? s0[g] : -1e30f;
            float v1 = (key0 + 16 <= sq) ? s1[g] : -1e30f;
            float rm = fmaxf(v0, v1);
            rm = fmaxf(rm, __shfl_xor(rm, 1));
            rm = fmaxf(rm, __shfl_xor(rm, 2));
            rm = fmaxf(rm, __shfl_xor(rm, 4));
            rm = fmaxf(rm, __shfl_xor(rm, 8));
            float mn = fmaxf(mrun[g], rm);
            float p0 = __expf(v0 - mn), p1 = __expf(v1 - mn);
            float rs = p0 + p1;
            rs += __shfl_xor(rs, 1); rs += __shfl_xor(rs, 2);
            rs += __shfl_xor(rs, 4); rs += __shfl_xor(rs, 8);
            float sc = __expf(mrun[g] - mn);
            lrun[g] = lrun[g] * sc + rs;
            mrun[g] = mn;
            p0a[g] = p0; p1a[g] = p1;
            #pragma unroll
            for (int j = 0; j < 4; ++j) o[j][g] *= sc;
        }

        #pragma unroll
        for (int g = 0; g < 8; ++g) {          // P: C-layout -> LDS -> A-layout
            int m = hi * 8 + g;
            Pl[wv][m][l16]      = f32_bf16(p0a[g]);
            Pl[wv][m][16 + l16] = f32_bf16(p1a[g]);
        }
        asm volatile("s_wait_dscnt 0x0" ::: "memory");
        FragBF pa;
        #pragma unroll
        for (int j = 0; j < 8; ++j) {
            pa.u[j]     = Pl[wv][l16][hi * 8 + j];
            pa.u[8 + j] = Pl[wv][l16][16 + hi * 8 + j];
        }
        #pragma unroll
        for (int nt = 0; nt < 4; ++nt) {       // P(16x32) * V(32x64)
            FragBF vb;
            const float* vp = qkv + (size_t)(b * TS + kc * 32 + hi * 8) * TQKV
                              + TD + TKVH * THD + kvh * THD + nt * 16 + l16;
            #pragma unroll
            for (int j = 0; j < 8; ++j) { vb.u[j] = f32_bf16(vp[(size_t)j * TQKV]); vb.u[8 + j] = f32_bf16(vp[(size_t)(16 + j) * TQKV]); }
            o[nt] = __builtin_amdgcn_wmma_f32_16x16x32_bf16(false, pa.v, false, vb.v, (short)0, o[nt], false, false);
        }
    }

    #pragma unroll
    for (int g = 0; g < 8; ++g) {
        const int sq = qt * 16 + hi * 8 + g;
        const float inv = 1.f / lrun[g];
        float* yp = y + (size_t)(b * TS + sq) * TD + h * THD;
        #pragma unroll
        for (int nt = 0; nt < 4; ++nt) yp[nt * 16 + l16] = o[nt][g] * inv;
    }
}

// ---------------- small kernels ----------------
__global__ void embed_kernel(const int* __restrict__ idx, const float* __restrict__ emb,
                             float* __restrict__ x)
{
    size_t i = (size_t)blockIdx.x * 256 + threadIdx.x;
    int t = (int)(i >> 10), d = (int)(i & 1023);
    x[i] = emb[(size_t)idx[t] * TD + d];
}

__global__ __launch_bounds__(256)
void rmsnorm_kernel(const float* __restrict__ x, const float* __restrict__ w,
                    float* __restrict__ out)
{
    const int t = blockIdx.x;
    const float* xr = x + (size_t)t * TD;
    float ss = 0.f;
    for (int d = threadIdx.x; d < TD; d += 256) { float v = xr[d]; ss += v * v; }
    for (int off = 16; off; off >>= 1) ss += __shfl_xor(ss, off);
    __shared__ float red[8];
    if ((threadIdx.x & 31) == 0) red[threadIdx.x >> 5] = ss;
    __syncthreads();
    float tot = 0.f;
    #pragma unroll
    for (int i = 0; i < 8; ++i) tot += red[i];
    const float inv = rsqrtf(tot * (1.0f / TD) + 1e-5f);
    float* orow = out + (size_t)t * TD;
    for (int d = threadIdx.x; d < TD; d += 256) orow[d] = xr[d] * inv * w[d];
}

__global__ void rope_kernel(float* __restrict__ qkv)
{
    int tid = blockIdx.x * 256 + threadIdx.x;
    int i = tid & 31; int rem = tid >> 5;
    int head = rem % (TH + TKVH); int t = rem / (TH + TKVH);
    int s = t & (TS - 1);
    float* p = qkv + (size_t)t * TQKV + (head < TH ? head * THD : TD + (head - TH) * THD);
    // freq = BASE^(-2i/HD), ln(1e6) = 13.815511
    float freq = __expf(-(float)(2 * i) * (1.0f / THD) * 13.815510557964274f);
    float ang = (float)s * freq;
    float c = cosf(ang), sn = sinf(ang);
    float x0 = p[2 * i], x1 = p[2 * i + 1];
    p[2 * i]     = x0 * c - x1 * sn;
    p[2 * i + 1] = x1 * c + x0 * sn;
}

__global__ __launch_bounds__(256)
void gate_kernel(const float* __restrict__ xn, const float* __restrict__ gw,
                 float* __restrict__ scores)
{
    const int t = blockIdx.x;
    const int e = threadIdx.x >> 5, lane = threadIdx.x & 31;
    const float* xr = xn + (size_t)t * TD;
    const float* gr = gw + (size_t)e * TD;
    float acc = 0.f;
    for (int d = lane; d < TD; d += 32) acc += xr[d] * gr[d];
    for (int off = 16; off; off >>= 1) acc += __shfl_xor(acc, off);
    if (lane == 0) scores[t * TE + e] = acc;
}

__global__ void router_kernel(const float* __restrict__ scores, float* __restrict__ wmat)
{
    int t = blockIdx.x * 256 + threadIdx.x;
    if (t >= TT) return;
    float s[TE]; float mx = -1e30f;
    #pragma unroll
    for (int e = 0; e < TE; ++e) { s[e] = scores[t * TE + e]; mx = fmaxf(mx, s[e]); }
    #pragma unroll
    for (int e = 0; e < TE; ++e) s[e] = __expf(s[e] - mx);
    int i0 = 0;
    #pragma unroll
    for (int e = 1; e < TE; ++e) if (s[e] > s[i0]) i0 = e;
    int i1 = (i0 == 0) ? 1 : 0;
    #pragma unroll
    for (int e = 0; e < TE; ++e) if (e != i0 && s[e] > s[i1]) i1 = e;
    float r = 1.f / (s[i0] + s[i1]);          // top-2 renormalized
    #pragma unroll
    for (int e = 0; e < TE; ++e) wmat[t * TE + e] = 0.f;
    wmat[t * TE + i0] = s[i0] * r;
    wmat[t * TE + i1] = s[i1] * r;
}

__global__ void scale_add_kernel(float* __restrict__ acc, const float* __restrict__ src,
                                 const float* __restrict__ wmat, int e)
{
    size_t i = (size_t)blockIdx.x * 256 + threadIdx.x;
    int t = (int)(i >> 10);
    acc[i] += wmat[t * TE + e] * src[i];
}

// =====================================================================
extern "C" void kernel_launch(void* const* d_in, const int* in_sizes, int n_in,
                              void* d_out, int out_size, void* d_ws, size_t ws_size,
                              hipStream_t stream)
{
    const int*   idx  = (const int*)  d_in[0];
    const float* temb = (const float*)d_in[1];
    const float* anw  = (const float*)d_in[2];
    const float* wqkv = (const float*)d_in[3];
    const float* wo   = (const float*)d_in[4];
    const float* fnw  = (const float*)d_in[5];
    const float* gw   = (const float*)d_in[6];
    const float* w1   = (const float*)d_in[7];
    const float* w2   = (const float*)d_in[8];
    const float* w3   = (const float*)d_in[9];
    const float* finw = (const float*)d_in[10];
    const float* outw = (const float*)d_in[11];
    float* logits = (float*)d_out;

    float* p = (float*)d_ws;
    float* xbuf = p; p += (size_t)TT * TD;
    float* hbuf = p; p += (size_t)TT * TD;
    float* xn   = p; p += (size_t)TT * TD;
    float* qkvb = p; p += (size_t)TT * TQKV;
    float* atty = p; p += (size_t)TT * TD;
    float* macc = p; p += (size_t)TT * TD;
    float* y13  = p; p += (size_t)TT * TI;
    float* tmpo = p; p += (size_t)TT * TD;
    float* scor = p; p += (size_t)TT * TE;
    float* wm   = p; p += (size_t)TT * TE;

    float* cur = xbuf;
    float* nxt = macc;

    embed_kernel<<<(TT * TD) / 256, 256, 0, stream>>>(idx, temb, cur);

    for (int l = 0; l < TLAY; ++l) {
        rmsnorm_kernel<<<TT, 256, 0, stream>>>(cur, anw + (size_t)l * TD, xn);
        gemm_bf16<<<dim3(TQKV / GN, TT / GM), 256, 0, stream>>>(
            xn, wqkv + (size_t)l * TQKV * TD, qkvb, nullptr, TT, TQKV, TD, 0);
        rope_kernel<<<(TT * (TH + TKVH) * 32) / 256, 256, 0, stream>>>(qkvb);
        attn_kernel<<<(TB * TH * (TS / 16)) / 4, dim3(32, 4), 0, stream>>>(qkvb, atty);
        gemm_bf16<<<dim3(TD / GN, TT / GM), 256, 0, stream>>>(
            atty, wo + (size_t)l * TD * TD, hbuf, cur, TT, TD, TD, 1);   // +residual

        rmsnorm_kernel<<<TT, 256, 0, stream>>>(hbuf, fnw + (size_t)l * TD, xn);
        gate_kernel<<<TT, 256, 0, stream>>>(xn, gw + (size_t)l * TE * TD, scor);
        router_kernel<<<(TT + 255) / 256, 256, 0, stream>>>(scor, wm);

        hipMemcpyAsync(nxt, hbuf, (size_t)TT * TD * sizeof(float),
                       hipMemcpyDeviceToDevice, stream);                 // acc = h
        for (int e = 0; e < TE; ++e) {
            const float* w1e = w1 + ((size_t)l * TE + e) * (size_t)TI * TD;
            const float* w3e = w3 + ((size_t)l * TE + e) * (size_t)TI * TD;
            const float* w2e = w2 + ((size_t)l * TE + e) * (size_t)TD * TI;
            gemm_bf16<<<dim3(TI / GN, TT / GM), 256, 0, stream>>>(
                xn, w1e, y13, nullptr, TT, TI, TD, 0);                   // y1
            gemm_bf16<<<dim3(TI / GN, TT / GM), 256, 0, stream>>>(
                xn, w3e, y13, y13, TT, TI, TD, 2);                       // silu(y1)*y3
            gemm_bf16<<<dim3(TD / GN, TT / GM), 256, 0, stream>>>(
                y13, w2e, tmpo, nullptr, TT, TD, TI, 0);                 // expert out
            scale_add_kernel<<<(TT * TD) / 256, 256, 0, stream>>>(nxt, tmpo, wm, e);
        }
        float* tsw = cur; cur = nxt; nxt = tsw;                          // x = h + moe
    }

    rmsnorm_kernel<<<TT, 256, 0, stream>>>(cur, finw, xn);
    gemm_bf16<<<dim3(TV / GN, TT / GM), 256, 0, stream>>>(
        xn, outw, logits, nullptr, TT, TV, TD, 0);
}
